// MultiHeadAttention_11166914970301
// MI455X (gfx1250) — compile-verified, matrix operand order checked by reading, not run on hardware
//
#include <hip/hip_runtime.h>
#include <hip/hip_bf16.h>
#include <math.h>

#define BB  32
#define TT  512
#define CC  1024
#define NHH 16
#define DHH 64
#define MM  (BB*TT)   // 16384

typedef __bf16 bf16_t;
typedef __attribute__((ext_vector_type(16))) __bf16 v16bf;
typedef __attribute__((ext_vector_type(8)))  __bf16 v8bf;
typedef __attribute__((ext_vector_type(4)))  __bf16 v4bf;
typedef __attribute__((ext_vector_type(8)))  float  v8f;

static __device__ __forceinline__ v16bf cat16(v8bf lo, v8bf hi) {
  return __builtin_shufflevector(lo, hi, 0,1,2,3,4,5,6,7,8,9,10,11,12,13,14,15);
}
static __device__ __forceinline__ v8f wmma_bf16(v16bf a, v16bf b, v8f c) {
  return __builtin_amdgcn_wmma_f32_16x16x32_bf16(false, a, false, b, (short)0, c, false, false);
}
// Fragment loader for the CDNA5 16-bit A/B layout:
// element j of the v16bf lives at k = k0 + (j/8)*16 + (lane>>4)*8 + (j%8).
// Caller passes base = row_ptr + k0 + khalf; we read [0..7] and [16..23].
static __device__ __forceinline__ v16bf ldfrag(const bf16_t* base) {
  v8bf lo = *(const v8bf*)(base);
  v8bf hi = *(const v8bf*)(base + 16);
  return cat16(lo, hi);
}

// ---------------------------------------------------------------- convert
__global__ void cvt_f32_bf16(const float* __restrict__ s, bf16_t* __restrict__ d, int n4) {
  int i = blockIdx.x * blockDim.x + threadIdx.x;
  if (i < n4) {
    float4 f = ((const float4*)s)[i];
    v4bf o = { (bf16_t)f.x, (bf16_t)f.y, (bf16_t)f.z, (bf16_t)f.w };
    ((v4bf*)d)[i] = o;
  }
}

// ---------------------------------------------------------------- QKV GEMM + RoPE
// grid = (CC/128, MM/128, 3); block = 256 (8 waves). Wave tile 32x64 (2x4 frags).
// mode 0/1 -> q/k with RoPE into (B,NH,T,DH); mode 2 -> v transposed into (B,NH,DH,T).
__global__ __launch_bounds__(256) void qkv_gemm_rope_kernel(
    const bf16_t* __restrict__ xb,
    const bf16_t* __restrict__ wq, const bf16_t* __restrict__ wk, const bf16_t* __restrict__ wv,
    bf16_t* __restrict__ qh, bf16_t* __restrict__ kh, bf16_t* __restrict__ vT)
{
  const int mode  = blockIdx.z;
  const bf16_t* w = (mode == 0) ? wq : (mode == 1 ? wk : wv);
  const int lane  = threadIdx.x & 31;
  const int wave  = threadIdx.x >> 5;
  const int lr    = lane & 15;
  const int half  = lane >> 4;
  const int khalf = half << 3;
  const int wm = wave & 3, wn = wave >> 2;
  const int m0 = blockIdx.y * 128 + wm * 32;
  const int n0 = blockIdx.x * 128 + wn * 64;

  const v8f vz = {0.f,0.f,0.f,0.f,0.f,0.f,0.f,0.f};
  v8f acc[2][4] = {{vz, vz, vz, vz}, {vz, vz, vz, vz}};

  const bf16_t* pa0 = xb + (size_t)(m0 +      lr) * CC + khalf;
  const bf16_t* pa1 = xb + (size_t)(m0 + 16 + lr) * CC + khalf;
  const bf16_t* pb0 = w  + (size_t)(n0 +      lr) * CC + khalf;
  const bf16_t* pb1 = w  + (size_t)(n0 + 16 + lr) * CC + khalf;
  const bf16_t* pb2 = w  + (size_t)(n0 + 32 + lr) * CC + khalf;
  const bf16_t* pb3 = w  + (size_t)(n0 + 48 + lr) * CC + khalf;

  // software-pipelined K loop: fragments for tile k+1 in flight during WMMAs of tile k
  v16bf A0 = ldfrag(pa0), A1 = ldfrag(pa1);
  v16bf B0 = ldfrag(pb0), B1 = ldfrag(pb1), B2 = ldfrag(pb2), B3 = ldfrag(pb3);
  for (int k0 = 32; k0 <= CC; k0 += 32) {
    v16bf nA0, nA1, nB0, nB1, nB2, nB3;
    const bool more = (k0 < CC);
    if (more) {
      nA0 = ldfrag(pa0 + k0); nA1 = ldfrag(pa1 + k0);
      nB0 = ldfrag(pb0 + k0); nB1 = ldfrag(pb1 + k0);
      nB2 = ldfrag(pb2 + k0); nB3 = ldfrag(pb3 + k0);
      __builtin_prefetch(pa0 + k0 + 32, 0, 1);     // gfx1250 global_prefetch_b8
      __builtin_prefetch(pb0 + k0 + 32, 0, 1);
    }
    acc[0][0] = wmma_bf16(A0, B0, acc[0][0]);
    acc[0][1] = wmma_bf16(A0, B1, acc[0][1]);
    acc[0][2] = wmma_bf16(A0, B2, acc[0][2]);
    acc[0][3] = wmma_bf16(A0, B3, acc[0][3]);
    acc[1][0] = wmma_bf16(A1, B0, acc[1][0]);
    acc[1][1] = wmma_bf16(A1, B1, acc[1][1]);
    acc[1][2] = wmma_bf16(A1, B2, acc[1][2]);
    acc[1][3] = wmma_bf16(A1, B3, acc[1][3]);
    if (more) { A0 = nA0; A1 = nA1; B0 = nB0; B1 = nB1; B2 = nB2; B3 = nB3; }
  }

  // C-frag element i: row = m0+mi*16 + i + 8*half, col = n0+ni*16 + lr
  for (int mi = 0; mi < 2; ++mi)
  for (int ni = 0; ni < 4; ++ni) {
    const int n  = n0 + ni * 16 + lr;
    const int h  = n >> 6;
    const int d  = n & 63;
    const float invf = __powf(10000.f, -(float)(d & ~1) * (1.0f / 64.0f));
    for (int i = 0; i < 8; ++i) {
      const int m = m0 + mi * 16 + 8 * half + i;
      const int b = m >> 9;       // / T
      const int t = m & 511;      // % T
      float val = acc[mi][ni][i];
      if (mode < 2) {
        float partner = __shfl_xor(val, 1, 32);  // paired element lives in lane^1
        float sth, cth;
        __sincosf((float)t * invf, &sth, &cth);
        float o = ((d & 1) == 0) ? (val * cth - partner * sth)
                                 : (partner * sth + val * cth);
        bf16_t* dst = (mode == 0) ? qh : kh;
        dst[(((size_t)(b * NHH + h)) * TT + t) * DHH + d] = (bf16_t)o;
      } else {
        vT[(((size_t)(b * NHH + h)) * DHH + d) * TT + t] = (bf16_t)val;
      }
    }
  }
}

// ---------------------------------------------------------------- flash attention
// grid = B*NH*(T/128); block = 256 (8 waves); each wave owns 16 query rows.
__global__ __launch_bounds__(256) void attn_kernel(
    const bf16_t* __restrict__ qh, const bf16_t* __restrict__ kh,
    const bf16_t* __restrict__ vT, bf16_t* __restrict__ ob)
{
  __shared__ __align__(16) bf16_t psh[8][16][40];  // wave-private P tiles, padded stride
  const int lane  = threadIdx.x & 31;
  const int wave  = threadIdx.x >> 5;
  const int lr    = lane & 15;
  const int half  = lane >> 4;
  const int khalf = half << 3;
  const int bh   = blockIdx.x >> 2;   // b*NH + h
  const int qblk = blockIdx.x & 3;
  const int b = bh >> 4;
  const int h = bh & 15;
  const int q0 = qblk * 128 + wave * 16;

  const bf16_t* qbase = qh + ((size_t)bh * TT + q0 + lr) * DHH + khalf;
  const v16bf qa0 = ldfrag(qbase);        // d in [0,32)
  const v16bf qa1 = ldfrag(qbase + 32);   // d in [32,64)

  const v8f vz = {0.f,0.f,0.f,0.f,0.f,0.f,0.f,0.f};
  v8f O0 = vz, O1 = vz, O2 = vz, O3 = vz;
  float mrow[8], lrow[8];
  for (int i = 0; i < 8; ++i) { mrow[i] = -INFINITY; lrow[i] = 0.f; }

  const bf16_t* kbase = kh + (size_t)bh * TT * DHH;
  const bf16_t* vbase = vT + (size_t)bh * DHH * TT;

  for (int kk0 = 0; kk0 < TT; kk0 += 32) {
    const bf16_t* kp0 = kbase + (size_t)(kk0 +      lr) * DHH + khalf;
    const bf16_t* kp1 = kbase + (size_t)(kk0 + 16 + lr) * DHH + khalf;
    v16bf kb00 = ldfrag(kp0);        v16bf kb01 = ldfrag(kp0 + 32);
    v16bf kb10 = ldfrag(kp1);        v16bf kb11 = ldfrag(kp1 + 32);

    // V^T fragments issued early: contiguous along keys thanks to (B,NH,DH,T) layout
    const bf16_t* vp = vbase + (size_t)lr * TT + kk0 + khalf;
    v16bf vb0 = ldfrag(vp);
    v16bf vb1 = ldfrag(vp + 16 * TT);
    v16bf vb2 = ldfrag(vp + 32 * TT);
    v16bf vb3 = ldfrag(vp + 48 * TT);

    v8f s0 = vz, s1 = vz;
    s0 = wmma_bf16(qa0, kb00, s0);   s0 = wmma_bf16(qa1, kb01, s0);
    s1 = wmma_bf16(qa0, kb10, s1);   s1 = wmma_bf16(qa1, kb11, s1);

    float p0[8], p1[8], alpha[8];
    for (int i = 0; i < 8; ++i) {
      float a = s0[i] * 0.125f;      // DH^-0.5
      float c = s1[i] * 0.125f;
      float r = fmaxf(a, c);
      for (int off = 1; off < 16; off <<= 1) r = fmaxf(r, __shfl_xor(r, off, 32));
      float nm = fmaxf(mrow[i], r);
      alpha[i] = __expf(mrow[i] - nm);
      mrow[i] = nm;
      p0[i] = __expf(a - nm);
      p1[i] = __expf(c - nm);
      float rs = p0[i] + p1[i];
      for (int off = 1; off < 16; off <<= 1) rs += __shfl_xor(rs, off, 32);
      lrow[i] = lrow[i] * alpha[i] + rs;
    }
    for (int i = 0; i < 8; ++i) {
      O0[i] *= alpha[i]; O1[i] *= alpha[i]; O2[i] *= alpha[i]; O3[i] *= alpha[i];
    }

    // C-layout P -> LDS -> A-layout fragment (wave-private, DS ops in-order per wave)
    for (int i = 0; i < 8; ++i) {
      psh[wave][i + 8 * half][lr]      = (bf16_t)p0[i];
      psh[wave][i + 8 * half][16 + lr] = (bf16_t)p1[i];
    }
    v16bf pa = ldfrag(&psh[wave][lr][khalf]);

    O0 = wmma_bf16(pa, vb0, O0);
    O1 = wmma_bf16(pa, vb1, O1);
    O2 = wmma_bf16(pa, vb2, O2);
    O3 = wmma_bf16(pa, vb3, O3);
  }

  bf16_t* obase = ob + (size_t)b * TT * CC + (size_t)h * DHH;
  for (int i = 0; i < 8; ++i) {
    const int t = q0 + i + 8 * half;
    const float il = 1.0f / lrow[i];
    const size_t ro = (size_t)t * CC;
    obase[ro +  0 + lr] = (bf16_t)(O0[i] * il);
    obase[ro + 16 + lr] = (bf16_t)(O1[i] * il);
    obase[ro + 32 + lr] = (bf16_t)(O2[i] * il);
    obase[ro + 48 + lr] = (bf16_t)(O3[i] * il);
  }
}

// ---------------------------------------------------------------- output projection
// grid = (CC/128, MM/128); block = 256 (8 waves). Wave tile 32x64, pipelined K loop.
__global__ __launch_bounds__(256) void oproj_kernel(
    const bf16_t* __restrict__ ob, const bf16_t* __restrict__ wo,
    float* __restrict__ out)
{
  const int lane  = threadIdx.x & 31;
  const int wave  = threadIdx.x >> 5;
  const int lr    = lane & 15;
  const int half  = lane >> 4;
  const int khalf = half << 3;
  const int wm = wave & 3, wn = wave >> 2;
  const int m0 = blockIdx.y * 128 + wm * 32;
  const int n0 = blockIdx.x * 128 + wn * 64;

  const v8f vz = {0.f,0.f,0.f,0.f,0.f,0.f,0.f,0.f};
  v8f acc[2][4] = {{vz, vz, vz, vz}, {vz, vz, vz, vz}};

  const bf16_t* pa0 = ob + (size_t)(m0 +      lr) * CC + khalf;
  const bf16_t* pa1 = ob + (size_t)(m0 + 16 + lr) * CC + khalf;
  const bf16_t* pb0 = wo + (size_t)(n0 +      lr) * CC + khalf;
  const bf16_t* pb1 = wo + (size_t)(n0 + 16 + lr) * CC + khalf;
  const bf16_t* pb2 = wo + (size_t)(n0 + 32 + lr) * CC + khalf;
  const bf16_t* pb3 = wo + (size_t)(n0 + 48 + lr) * CC + khalf;

  v16bf A0 = ldfrag(pa0), A1 = ldfrag(pa1);
  v16bf B0 = ldfrag(pb0), B1 = ldfrag(pb1), B2 = ldfrag(pb2), B3 = ldfrag(pb3);
  for (int k0 = 32; k0 <= CC; k0 += 32) {
    v16bf nA0, nA1, nB0, nB1, nB2, nB3;
    const bool more = (k0 < CC);
    if (more) {
      nA0 = ldfrag(pa0 + k0); nA1 = ldfrag(pa1 + k0);
      nB0 = ldfrag(pb0 + k0); nB1 = ldfrag(pb1 + k0);
      nB2 = ldfrag(pb2 + k0); nB3 = ldfrag(pb3 + k0);
      __builtin_prefetch(pa0 + k0 + 32, 0, 1);
      __builtin_prefetch(pb0 + k0 + 32, 0, 1);
    }
    acc[0][0] = wmma_bf16(A0, B0, acc[0][0]);
    acc[0][1] = wmma_bf16(A0, B1, acc[0][1]);
    acc[0][2] = wmma_bf16(A0, B2, acc[0][2]);
    acc[0][3] = wmma_bf16(A0, B3, acc[0][3]);
    acc[1][0] = wmma_bf16(A1, B0, acc[1][0]);
    acc[1][1] = wmma_bf16(A1, B1, acc[1][1]);
    acc[1][2] = wmma_bf16(A1, B2, acc[1][2]);
    acc[1][3] = wmma_bf16(A1, B3, acc[1][3]);
    if (more) { A0 = nA0; A1 = nA1; B0 = nB0; B1 = nB1; B2 = nB2; B3 = nB3; }
  }

  for (int mi = 0; mi < 2; ++mi)
  for (int ni = 0; ni < 4; ++ni) {
    const int n = n0 + ni * 16 + lr;
    for (int i = 0; i < 8; ++i) {
      const int m = m0 + mi * 16 + 8 * half + i;
      out[(size_t)m * CC + n] = acc[mi][ni][i];
    }
  }
}

// ---------------------------------------------------------------- launch
extern "C" void kernel_launch(void* const* d_in, const int* in_sizes, int n_in,
                              void* d_out, int out_size, void* d_ws, size_t ws_size,
                              hipStream_t stream) {
  (void)in_sizes; (void)n_in; (void)out_size; (void)ws_size;
  const float* x  = (const float*)d_in[0];
  const float* Wq = (const float*)d_in[1];
  const float* Wk = (const float*)d_in[2];
  const float* Wv = (const float*)d_in[3];
  const float* Wo = (const float*)d_in[4];

  char* p = (char*)d_ws;
  bf16_t* xb  = (bf16_t*)p; p += (size_t)MM * CC * 2;   // 32 MB (re-used as attention output)
  bf16_t* wqb = (bf16_t*)p; p += (size_t)CC * CC * 2;
  bf16_t* wkb = (bf16_t*)p; p += (size_t)CC * CC * 2;
  bf16_t* wvb = (bf16_t*)p; p += (size_t)CC * CC * 2;
  bf16_t* wob = (bf16_t*)p; p += (size_t)CC * CC * 2;
  bf16_t* qhb = (bf16_t*)p; p += (size_t)MM * CC * 2;   // (B,NH,T,DH)
  bf16_t* khb = (bf16_t*)p; p += (size_t)MM * CC * 2;   // (B,NH,T,DH)
  bf16_t* vtb = (bf16_t*)p; p += (size_t)MM * CC * 2;   // (B,NH,DH,T)
  bf16_t* obb = xb;                                     // alias: x no longer needed post-QKV

  int n4 = MM * CC / 4;
  cvt_f32_bf16<<<(n4 + 255) / 256, 256, 0, stream>>>(x, xb, n4);
  n4 = CC * CC / 4;
  cvt_f32_bf16<<<(n4 + 255) / 256, 256, 0, stream>>>(Wq, wqb, n4);
  cvt_f32_bf16<<<(n4 + 255) / 256, 256, 0, stream>>>(Wk, wkb, n4);
  cvt_f32_bf16<<<(n4 + 255) / 256, 256, 0, stream>>>(Wv, wvb, n4);
  cvt_f32_bf16<<<(n4 + 255) / 256, 256, 0, stream>>>(Wo, wob, n4);

  qkv_gemm_rope_kernel<<<dim3(CC / 128, MM / 128, 3), 256, 0, stream>>>(
      xb, wqb, wkb, wvb, qhb, khb, vtb);

  attn_kernel<<<dim3(BB * NHH * (TT / 128)), 256, 0, stream>>>(qhb, khb, vtb, obb);

  oproj_kernel<<<dim3(CC / 128, MM / 128), 256, 0, stream>>>(obb, wob, (float*)d_out);
}
